// Segmenter_4028679324111
// MI455X (gfx1250) — matrix operand order repaired and checked by reading, not compiled, not run on hardware
//
#include <hip/hip_runtime.h>

#define DEV __device__ __forceinline__

typedef __attribute__((ext_vector_type(16))) __bf16 v16bf;
typedef __attribute__((ext_vector_type(8)))  float  v8f;
typedef unsigned int   u32;
typedef unsigned short u16;
typedef __attribute__((ext_vector_type(4))) u32 u32x4;

union Frag { v16bf v; u32x4 q[2]; };

constexpr int   NN      = 8192;
constexpr int   DM      = 768;
constexpr int   DP      = 512;
constexpr int   NC      = 128;
constexpr float T_SCALE = 10.0f;
constexpr float ALPHA   = 0.02f;
constexpr float MOM     = 0.9f;
constexpr float EPSN    = 1e-12f;

DEV u16 f2bf(float x) {
  u32 u = __float_as_uint(x);
  u = (u + 0x7FFFu + ((u >> 16) & 1u)) >> 16;
  return (u16)u;
}

DEV v8f zero8() {
  v8f z;
  #pragma unroll
  for (int i = 0; i < 8; ++i) z[i] = 0.f;
  return z;
}

DEV v8f wmma_bf16(v16bf a, v16bf b, v8f c) {
  return __builtin_amdgcn_wmma_f32_16x16x32_bf16(false, a, false, b, (short)0, c, false, false);
}

// ---------------- init: zero atomically-accumulated buffers -----------------
__global__ void k_init(float* __restrict__ z, int n, float* __restrict__ out2) {
  int idx = blockIdx.x * 256 + threadIdx.x;
  if (idx < n) z[idx] = 0.f;
  if (idx < 2) out2[idx] = 0.f;
}

// ---------------- Psi row-normalize + Psi_l^T bf16 --------------------------
__global__ __launch_bounds__(128) void k_norm_psi(const float* __restrict__ Psi,
                                                  float* __restrict__ Psi_n,
                                                  u16*   __restrict__ PsiLT) {
  __shared__ float red[128];
  const int i = blockIdx.x, t = threadIdx.x;
  float vals[4]; float ss = 0.f;
  #pragma unroll
  for (int e = 0; e < 4; ++e) {
    int c = t + e * 128;
    vals[e] = Psi[(size_t)i * DP + c];
    ss += vals[e] * vals[e];
  }
  red[t] = ss; __syncthreads();
  for (int s = 64; s > 0; s >>= 1) { if (t < s) red[t] += red[t + s]; __syncthreads(); }
  const float inv = 1.f / fmaxf(sqrtf(red[0]), EPSN);
  const float pls = T_SCALE / (float)NN;
  #pragma unroll
  for (int e = 0; e < 4; ++e) {
    int c = t + e * 128;
    Psi_n[(size_t)i * DP + c] = vals[e] * inv;
    PsiLT[(size_t)c * NN + i] = f2bf(vals[e] * pls);
  }
}

// ---------------- FF row-normalize -> bf16 ----------------------------------
__global__ __launch_bounds__(256) void k_norm_ff(const float* __restrict__ FF,
                                                 u16* __restrict__ FFn) {
  __shared__ float red[256];
  const int i = blockIdx.x, t = threadIdx.x;
  float vals[3]; float ss = 0.f;
  #pragma unroll
  for (int e = 0; e < 3; ++e) {
    int c = t + e * 256;
    vals[e] = FF[(size_t)i * DM + c];
    ss += vals[e] * vals[e];
  }
  red[t] = ss; __syncthreads();
  for (int s = 128; s > 0; s >>= 1) { if (t < s) red[t] += red[t + s]; __syncthreads(); }
  const float inv = 1.f / fmaxf(sqrtf(red[0]), EPSN);
  #pragma unroll
  for (int e = 0; e < 3; ++e) {
    int c = t + e * 256;
    FFn[(size_t)i * DM + c] = f2bf(vals[e] * inv);
  }
}

// ---------------- center squared norms --------------------------------------
__global__ void k_csq(const float* __restrict__ C, float* __restrict__ csq) {
  int t = threadIdx.x;
  if (t < NC) {
    float s = 0.f;
    for (int k = 0; k < DP; ++k) { float v = C[(size_t)t * DP + k]; s = fmaf(v, v, s); }
    csq[t] = s;
  }
}

// ---------------- logits + argmax + cluster accumulation --------------------
__global__ __launch_bounds__(128) void k_logits(const float* __restrict__ Psi_n,
                                                const float* __restrict__ C,
                                                const float* __restrict__ csq,
                                                float* __restrict__ logits,
                                                float* __restrict__ counts,
                                                float* __restrict__ csum) {
  __shared__ float prow[DP];
  __shared__ float red[128];
  __shared__ int   redi[128];
  const int i = blockIdx.x, t = threadIdx.x;
  float ss = 0.f;
  for (int c = t; c < DP; c += 128) {
    float v = Psi_n[(size_t)i * DP + c];
    prow[c] = v; ss += v * v;
  }
  red[t] = ss; __syncthreads();
  for (int s = 64; s > 0; s >>= 1) { if (t < s) red[t] += red[t + s]; __syncthreads(); }
  const float psq = red[0];
  __syncthreads();
  float dot = 0.f;
  const float* cj = C + (size_t)t * DP;
  for (int k = 0; k < DP; ++k) dot = fmaf(prow[k], cj[k], dot);
  const float lg = -(psq + csq[t] - 2.f * dot);
  logits[(size_t)i * NC + t] = lg;
  red[t] = lg; redi[t] = t; __syncthreads();
  for (int s = 64; s > 0; s >>= 1) {
    if (t < s) {
      float o = red[t + s]; int oi = redi[t + s];
      if (o > red[t] || (o == red[t] && oi < redi[t])) { red[t] = o; redi[t] = oi; }
    }
    __syncthreads();
  }
  const int a = redi[0];
  if (t == 0) atomicAdd(&counts[a], 1.f);
  for (int c = t; c < DP; c += 128) atomicAdd(&csum[(size_t)a * DP + c], prow[c]);
}

// ---------------- EMA centroid update ---------------------------------------
__global__ void k_new_centers(const float* __restrict__ C,
                              const float* __restrict__ counts,
                              const float* __restrict__ csum,
                              float* __restrict__ outC) {
  int idx = blockIdx.x * 256 + threadIdx.x;
  if (idx >= NC * DP) return;
  int cls = idx >> 9;
  float cnt = fmaxf(counts[cls], 1.f);
  outC[idx] = C[idx] * MOM + (csum[idx] / cnt) * (1.f - MOM);
}

// ---------------- pass 1: row sums of clamped A (WMMA bf16) -----------------
__global__ __launch_bounds__(256) void k_arowsum(const u16* __restrict__ FFn,
                                                 float* __restrict__ Dacc) {
  const int bi = blockIdx.x, bj = blockIdx.y;
  const int w = threadIdx.x >> 5, lane = threadIdx.x & 31;
  const int wr = w & 3, wc = w >> 2;
  const int half = lane >> 4, rl = lane & 15;
  const int rbase = bi * 128 + wr * 32;
  const int cbase = bj * 128 + wc * 64;
  v8f acc[8];
  #pragma unroll
  for (int s = 0; s < 8; ++s) acc[s] = zero8();

  for (int k0 = 0; k0 < DM; k0 += 32) {
    Frag a[2], b[4];
    #pragma unroll
    for (int sr = 0; sr < 2; ++sr) {
      const u16* p = FFn + (size_t)(rbase + sr * 16 + rl) * DM + k0 + half * 8;
      a[sr].q[0] = *(const u32x4*)p;
      a[sr].q[1] = *(const u32x4*)(p + 16);
    }
    #pragma unroll
    for (int sc = 0; sc < 4; ++sc) {
      const u16* p = FFn + (size_t)(cbase + sc * 16 + rl) * DM + k0 + half * 8;
      b[sc].q[0] = *(const u32x4*)p;
      b[sc].q[1] = *(const u32x4*)(p + 16);
    }
    #pragma unroll
    for (int sr = 0; sr < 2; ++sr)
      #pragma unroll
      for (int sc = 0; sc < 4; ++sc)
        acc[sr * 4 + sc] = wmma_bf16(a[sr].v, b[sc].v, acc[sr * 4 + sc]);
  }

  #pragma unroll
  for (int sr = 0; sr < 2; ++sr) {
    float rp[8];
    #pragma unroll
    for (int r = 0; r < 8; ++r) rp[r] = 0.f;
    #pragma unroll
    for (int sc = 0; sc < 4; ++sc) {
      const int gc = cbase + sc * 16 + rl;
      #pragma unroll
      for (int r = 0; r < 8; ++r) {
        const int gr = rbase + sr * 16 + half * 8 + r;
        float v = acc[sr * 4 + sc][r];
        v = (gr == gc) ? 0.f : fmaxf(v, 0.f);
        rp[r] += v;
      }
    }
    #pragma unroll
    for (int m = 1; m < 16; m <<= 1) {
      #pragma unroll
      for (int r = 0; r < 8; ++r) rp[r] += __shfl_xor(rp[r], m, 32);
    }
    if (rl == 0) {
      #pragma unroll
      for (int r = 0; r < 8; ++r)
        atomicAdd(&Dacc[rbase + sr * 16 + half * 8 + r], rp[r]);
    }
  }
}

// ---------------- Drs = rsqrt(D / (N-1)) ------------------------------------
__global__ void k_drs(const float* __restrict__ Dacc, float* __restrict__ Drs) {
  int idx = blockIdx.x * 256 + threadIdx.x;
  if (idx < NN) Drs[idx] = rsqrtf(Dacc[idx] * (1.0f / (float)(NN - 1)));
}

// ---------------- pass 2: gPsi = gram @ Psi_l (double WMMA) -----------------
#define GSTR 144
__global__ __launch_bounds__(256) void k_gpsi(const u16* __restrict__ FFn,
                                              const u16* __restrict__ PsiLT,
                                              const float* __restrict__ Drs,
                                              float* __restrict__ gPsi) {
  __shared__ u16   gram[128 * GSTR];
  __shared__ float drow[128];
  __shared__ float dcol[128];
  const int bi = blockIdx.x, bc = blockIdx.y;
  const int t = threadIdx.x;
  const int w = t >> 5, lane = t & 31;
  const int wr = w & 3, wc = w >> 2;
  const int half = lane >> 4, rl = lane & 15;
  const int rbase = wr * 32;   // local row base of this wave's tile
  const int cbase = wc * 64;   // local col base

  if (t < 128) drow[t] = Drs[bi * 128 + t];

  v8f g[8];
  #pragma unroll
  for (int s = 0; s < 8; ++s) g[s] = zero8();

  for (int kb = 0; kb < 64; ++kb) {
    if (t < 128) dcol[t] = Drs[kb * 128 + t];
    __syncthreads();   // dcol/drow ready; gram free of prior readers

    // --- recompute A tile (rows bi*128.., cols kb*128..) from FF_n ---
    v8f a8[8];
    #pragma unroll
    for (int s = 0; s < 8; ++s) a8[s] = zero8();
    for (int k0 = 0; k0 < DM; k0 += 32) {
      Frag a[2], b[4];
      #pragma unroll
      for (int sr = 0; sr < 2; ++sr) {
        const u16* p = FFn + (size_t)(bi * 128 + rbase + sr * 16 + rl) * DM + k0 + half * 8;
        a[sr].q[0] = *(const u32x4*)p;
        a[sr].q[1] = *(const u32x4*)(p + 16);
      }
      #pragma unroll
      for (int sc = 0; sc < 4; ++sc) {
        const u16* p = FFn + (size_t)(kb * 128 + cbase + sc * 16 + rl) * DM + k0 + half * 8;
        b[sc].q[0] = *(const u32x4*)p;
        b[sc].q[1] = *(const u32x4*)(p + 16);
      }
      #pragma unroll
      for (int sr = 0; sr < 2; ++sr)
        #pragma unroll
        for (int sc = 0; sc < 4; ++sc)
          a8[sr * 4 + sc] = wmma_bf16(a[sr].v, b[sc].v, a8[sr * 4 + sc]);
    }

    // --- clamp / diag-zero / Drs-scale -> bf16 gram tile in LDS ---
    #pragma unroll
    for (int sr = 0; sr < 2; ++sr)
      #pragma unroll
      for (int sc = 0; sc < 4; ++sc) {
        const int lc = cbase + sc * 16 + rl;
        const int gk = kb * 128 + lc;
        #pragma unroll
        for (int r = 0; r < 8; ++r) {
          const int lr = rbase + sr * 16 + half * 8 + r;
          const int gi = bi * 128 + lr;
          float v = a8[sr * 4 + sc][r];
          v = (gi == gk) ? 0.f : fmaxf(v, 0.f);
          v *= drow[lr] * dcol[lc];
          gram[lr * GSTR + lc] = f2bf(v);
        }
      }
    __syncthreads();

    // --- gacc += gram(128x128) x Psi_l[kb-tile](128x128) ---
    #pragma unroll
    for (int ks = 0; ks < 4; ++ks) {
      Frag a[2], b[4];
      #pragma unroll
      for (int sr = 0; sr < 2; ++sr) {
        const u16* p = &gram[(rbase + sr * 16 + rl) * GSTR + ks * 32 + half * 8];
        a[sr].q[0] = *(const u32x4*)p;
        a[sr].q[1] = *(const u32x4*)(p + 16);
      }
      #pragma unroll
      for (int sc = 0; sc < 4; ++sc) {
        const int n = bc * 128 + cbase + sc * 16 + rl;
        const u16* p = PsiLT + (size_t)n * NN + kb * 128 + ks * 32 + half * 8;
        b[sc].q[0] = *(const u32x4*)p;
        b[sc].q[1] = *(const u32x4*)(p + 16);
      }
      #pragma unroll
      for (int sr = 0; sr < 2; ++sr)
        #pragma unroll
        for (int sc = 0; sc < 4; ++sc)
          g[sr * 4 + sc] = wmma_bf16(a[sr].v, b[sc].v, g[sr * 4 + sc]);
    }
    __syncthreads();   // protect gram/dcol before next kb
  }

  // --- write gPsi tile ---
  #pragma unroll
  for (int sr = 0; sr < 2; ++sr)
    #pragma unroll
    for (int sc = 0; sc < 4; ++sc) {
      const int gc = bc * 128 + cbase + sc * 16 + rl;
      #pragma unroll
      for (int r = 0; r < 8; ++r) {
        const int gr = bi * 128 + rbase + sr * 16 + half * 8 + r;
        gPsi[(size_t)gr * DP + gc] = g[sr * 4 + sc][r];
      }
    }
}

// ---------------- R = Psi_l^T gPsi -> loss & reg ----------------------------
__global__ __launch_bounds__(256) void k_rred(const float* __restrict__ Psi,
                                              const float* __restrict__ gPsi,
                                              float* __restrict__ out2) {
  __shared__ float ps[64][16];
  __shared__ float gp[64][16];
  __shared__ float red[256];
  const int bi = blockIdx.x, bj = blockIdx.y, t = threadIdx.x;
  const int ii = t >> 4, jj = t & 15;
  const float pls = T_SCALE / (float)NN;
  float acc = 0.f;
  for (int k0 = 0; k0 < NN; k0 += 64) {
    for (int e = t; e < 1024; e += 256) {
      int kk = e >> 4, cc = e & 15;
      ps[kk][cc] = Psi[(size_t)(k0 + kk) * DP + bi * 16 + cc] * pls;
      gp[kk][cc] = gPsi[(size_t)(k0 + kk) * DP + bj * 16 + cc];
    }
    __syncthreads();
    #pragma unroll 8
    for (int kk = 0; kk < 64; ++kk) acc = fmaf(ps[kk][ii], gp[kk][jj], acc);
    __syncthreads();
  }
  const int gi = bi * 16 + ii, gj = bj * 16 + jj;
  float lossC = (gi == gj) ? -acc : 0.f;
  float regC  = (gj > gi) ? ALPHA * acc * acc : 0.f;
  red[t] = lossC; __syncthreads();
  for (int s = 128; s > 0; s >>= 1) { if (t < s) red[t] += red[t + s]; __syncthreads(); }
  if (t == 0) atomicAdd(&out2[0], red[0]);
  __syncthreads();
  red[t] = regC; __syncthreads();
  for (int s = 128; s > 0; s >>= 1) { if (t < s) red[t] += red[t + s]; __syncthreads(); }
  if (t == 0) atomicAdd(&out2[1], red[0]);
}

// ---------------------------------------------------------------------------
extern "C" void kernel_launch(void* const* d_in, const int* in_sizes, int n_in,
                              void* d_out, int out_size, void* d_ws, size_t ws_size,
                              hipStream_t stream) {
  (void)in_sizes; (void)n_in; (void)out_size; (void)ws_size;
  const float* FF  = (const float*)d_in[0];
  const float* Psi = (const float*)d_in[1];
  const float* C   = (const float*)d_in[2];
  float* out = (float*)d_out;
  float* out_logits  = out;
  float* out_centers = out + (size_t)NN * NC;
  float* out2        = out + (size_t)NN * NC + (size_t)NC * DP;

  char* w = (char*)d_ws;
  float* Psi_n = (float*)w; w += (size_t)NN * DP * 4;
  u16*   FFn   = (u16*)w;   w += (size_t)NN * DM * 2;
  u16*   PsiLT = (u16*)w;   w += (size_t)DP * NN * 2;
  float* gPsi  = (float*)w; w += (size_t)NN * DP * 4;
  float* counts = (float*)w; w += NC * 4;              // | contiguous zeroed
  float* csum   = (float*)w; w += (size_t)NC * DP * 4; // | region starts at
  float* Dacc   = (float*)w; w += NN * 4;              // | counts
  float* Drs    = (float*)w; w += NN * 4;
  float* csq    = (float*)w; w += NC * 4;

  const int nzero = NC + NC * DP + NN;  // counts + csum + Dacc
  k_init<<<(nzero + 255) / 256, 256, 0, stream>>>(counts, nzero, out2);
  k_norm_psi<<<NN, 128, 0, stream>>>(Psi, Psi_n, PsiLT);
  k_norm_ff<<<NN, 256, 0, stream>>>(FF, FFn);
  k_csq<<<1, 128, 0, stream>>>(C, csq);
  k_logits<<<NN, 128, 0, stream>>>(Psi_n, C, csq, out_logits, counts, csum);
  k_new_centers<<<(NC * DP + 255) / 256, 256, 0, stream>>>(C, counts, csum, out_centers);
  k_arowsum<<<dim3(64, 64), 256, 0, stream>>>(FFn, Dacc);
  k_drs<<<(NN + 255) / 256, 256, 0, stream>>>(Dacc, Drs);
  k_gpsi<<<dim3(64, 4), 256, 0, stream>>>(FFn, PsiLT, Drs, gPsi);
  k_rred<<<dim3(32, 32), 256, 0, stream>>>(Psi, gPsi, out2);
}
